// SIGRegVectorized_9457517986137
// MI455X (gfx1250) — matrix operand order repaired and checked by reading, not compiled, not run on hardware
//
#include <hip/hip_runtime.h>
#include <hip/hip_bf16.h>
#include <math.h>

typedef __attribute__((ext_vector_type(16))) __bf16 bf16x16;
typedef __attribute__((ext_vector_type(8)))  __bf16 bf16x8;
typedef __attribute__((ext_vector_type(4)))  __bf16 bf16x4;
typedef __attribute__((ext_vector_type(8)))  float  f32x8;

union FragU { bf16x16 v; bf16x8 h[2]; };

constexpr int Brows = 8192;   // batch
constexpr int Ddim  = 2048;   // feature dim (GEMM K)
constexpr int Kcols = 512;    // directions (GEMM N)

constexpr int BM = 128, BN = 128, BKt = 32;
constexpr int LDK = 40;       // padded LDS row stride (bf16 elems)
constexpr int NT  = 17;       // Epps-Pulley points

// ---------------------------------------------------------------- zero out
__global__ void sig_zero_kernel(float* out) {
    if (threadIdx.x == 0 && blockIdx.x == 0) out[0] = 0.0f;
}

// ------------------------------------------------- column norms of dirs
__global__ __launch_bounds__(256) void sig_colnorm_kernel(
    const float* __restrict__ dirs, float* __restrict__ invn) {
    __shared__ float red[8][32];
    const int kloc  = threadIdx.x & 31;
    const int slice = threadIdx.x >> 5;
    const int k = blockIdx.x * 32 + kloc;
    float s = 0.0f;
    for (int d = slice; d < Ddim; d += 8) {
        float x = dirs[(size_t)d * Kcols + k];
        s = fmaf(x, x, s);
    }
    red[slice][kloc] = s;
    __syncthreads();
    if (slice == 0) {
        float t = 0.0f;
        #pragma unroll
        for (int i = 0; i < 8; ++i) t += red[i][kloc];
        float nrm = sqrtf(t);
        invn[k] = 1.0f / fmaxf(nrm, 1e-12f);
    }
}

// ------------------------------------------------- split-bf16 WMMA GEMM
// proj = emb(8192x2048) @ (dirs * invn)(2048x512), fp32-accurate via
// 3-term bf16 split: hi*hi + hi*lo + lo*hi.
// Register-staged double buffering: next tile's global loads issue under
// the current tile's WMMAs.
__global__ __launch_bounds__(256) void sig_gemm_kernel(
    const float* __restrict__ A,     // Brows x Ddim
    const float* __restrict__ Dirs,  // Ddim x Kcols
    const float* __restrict__ invn,  // Kcols
    float* __restrict__ Cout) {      // Brows x Kcols
    __shared__ __bf16 sA[2][BM][LDK];   // [hi/lo][m][k]
    __shared__ __bf16 sB[2][BN][LDK];   // [hi/lo][n][k] (n-major)

    const int tid  = threadIdx.x;
    const int lane = tid & 31;
    const int wave = tid >> 5;
    const int blockM = blockIdx.x * BM;
    const int blockN = blockIdx.y * BN;
    const int waveM = (wave & 3) * 32;   // 4 waves along M
    const int waveN = (wave >> 2) * 64;  // 2 waves along N

    f32x8 acc[2][4];
    #pragma unroll
    for (int mi = 0; mi < 2; ++mi)
        #pragma unroll
        for (int ni = 0; ni < 4; ++ni) acc[mi][ni] = 0.0f;

    // thread slots in the staging loaders
    const int aRow = tid >> 3;            // + i*32  (A: 4 rows, same kq)
    const int aKq  = (tid & 7) * 4;
    const int bN4  = (tid & 31) * 4;      // B: 4 consecutive k-rows, 4 cols
    const int bKr0 = (tid >> 5) * 4;

    // per-thread column scales (fold normalize into the GEMM)
    const float4 scl = *(const float4*)&invn[blockN + bN4];
    const float sclv[4] = {scl.x, scl.y, scl.z, scl.w};

    float4 regA[4], regB[4];
    #pragma unroll
    for (int i = 0; i < 4; ++i) {
        regA[i] = *(const float4*)&A[(size_t)(blockM + aRow + i * 32) * Ddim + aKq];
        regB[i] = *(const float4*)&Dirs[(size_t)(bKr0 + i) * Kcols + blockN + bN4];
    }

    for (int k0 = 0; k0 < Ddim; k0 += BKt) {
        __syncthreads();   // previous tile's fragment reads are done

        // ---- stage A tile from regs (128x32 fp32 -> bf16 hi/lo)
        #pragma unroll
        for (int i = 0; i < 4; ++i) {
            const float xs[4] = {regA[i].x, regA[i].y, regA[i].z, regA[i].w};
            bf16x4 hi, lo;
            #pragma unroll
            for (int j = 0; j < 4; ++j) {
                __bf16 h = (__bf16)xs[j];
                hi[j] = h;
                lo[j] = (__bf16)(xs[j] - (float)h);
            }
            *(bf16x4*)&sA[0][aRow + i * 32][aKq] = hi;
            *(bf16x4*)&sA[1][aRow + i * 32][aKq] = lo;
        }
        // ---- stage B tile from regs (scaled, transposed; k packs as b64)
        {
            float bxs[4][4];
            #pragma unroll
            for (int i = 0; i < 4; ++i) {
                bxs[i][0] = regB[i].x; bxs[i][1] = regB[i].y;
                bxs[i][2] = regB[i].z; bxs[i][3] = regB[i].w;
            }
            #pragma unroll
            for (int j = 0; j < 4; ++j) {
                bf16x4 hi, lo;
                #pragma unroll
                for (int i = 0; i < 4; ++i) {
                    const float x = bxs[i][j] * sclv[j];
                    __bf16 h = (__bf16)x;
                    hi[i] = h;
                    lo[i] = (__bf16)(x - (float)h);
                }
                *(bf16x4*)&sB[0][bN4 + j][bKr0] = hi;
                *(bf16x4*)&sB[1][bN4 + j][bKr0] = lo;
            }
        }
        __syncthreads();   // staged tile visible to all waves

        // ---- prefetch next tile into registers (hidden under WMMAs)
        const int kn = k0 + BKt;
        if (kn < Ddim) {
            #pragma unroll
            for (int i = 0; i < 4; ++i) {
                regA[i] = *(const float4*)&A[(size_t)(blockM + aRow + i * 32) * Ddim + kn + aKq];
                regB[i] = *(const float4*)&Dirs[(size_t)(kn + bKr0 + i) * Kcols + blockN + bN4];
            }
        }
        // ---- deep L2 prefetch two tiles ahead (global_prefetch_b8, free)
        if (k0 + 2 * BKt < Ddim) {
            __builtin_prefetch(&A[(size_t)(blockM + aRow) * Ddim + k0 + 2 * BKt + aKq], 0, 0);
            __builtin_prefetch(&Dirs[(size_t)(k0 + 2 * BKt + bKr0) * Kcols + blockN + bN4], 0, 0);
        }

        // ---- A fragments: ISA 16-bit A layout: K chunks {kh..kh+7, 16+kh..}
        const int kh = (lane >> 4) * 8;
        bf16x16 aF[2][2];
        #pragma unroll
        for (int mi = 0; mi < 2; ++mi) {
            const int m = waveM + mi * 16 + (lane & 15);
            #pragma unroll
            for (int vv = 0; vv < 2; ++vv) {
                const __bf16* p = &sA[vv][m][0];
                FragU u;
                u.h[0] = *(const bf16x8*)(p + kh);
                u.h[1] = *(const bf16x8*)(p + 16 + kh);
                aF[vv][mi] = u.v;
            }
        }
        // ---- B fragments: ISA 16-bit B layout: K chunks {kb..kb+7, 16+kb..}
        const int kb = (lane >> 4) * 8;
        #pragma unroll
        for (int ni = 0; ni < 4; ++ni) {
            const int n = waveN + ni * 16 + (lane & 15);
            bf16x16 bHi, bLo;
            {
                const __bf16* p = &sB[0][n][kb];
                FragU u; u.h[0] = *(const bf16x8*)p; u.h[1] = *(const bf16x8*)(p + 16);
                bHi = u.v;
            }
            {
                const __bf16* p = &sB[1][n][kb];
                FragU u; u.h[0] = *(const bf16x8*)p; u.h[1] = *(const bf16x8*)(p + 16);
                bLo = u.v;
            }
            #pragma unroll
            for (int mi = 0; mi < 2; ++mi) {
                // small terms first, then dominant hi*hi
                acc[mi][ni] = __builtin_amdgcn_wmma_f32_16x16x32_bf16(
                    false, aF[1][mi], false, bHi, (short)0, acc[mi][ni], false, false);
                acc[mi][ni] = __builtin_amdgcn_wmma_f32_16x16x32_bf16(
                    false, aF[0][mi], false, bLo, (short)0, acc[mi][ni], false, false);
                acc[mi][ni] = __builtin_amdgcn_wmma_f32_16x16x32_bf16(
                    false, aF[0][mi], false, bHi, (short)0, acc[mi][ni], false, false);
            }
        }
    }

    // ---- epilogue: C layout — VGPR r holds M=r (lanes 0-15) / M=8+r (16-31)
    #pragma unroll
    for (int mi = 0; mi < 2; ++mi)
        #pragma unroll
        for (int ni = 0; ni < 4; ++ni) {
            const int n = blockN + waveN + ni * 16 + (lane & 15);
            const int mbase = blockM + waveM + mi * 16 + ((lane >> 4) << 3);
            #pragma unroll
            for (int r = 0; r < 8; ++r)
                Cout[(size_t)(mbase + r) * Kcols + n] = acc[mi][ni][r];
        }
}

// ------------------------------------------------- column mean / std(ddof=1)
__global__ __launch_bounds__(256) void sig_colstats_kernel(
    const float* __restrict__ P, float* __restrict__ mu, float* __restrict__ rstd) {
    __shared__ float redS[8][32], redQ[8][32];
    const int kloc  = threadIdx.x & 31;
    const int slice = threadIdx.x >> 5;
    const int k = blockIdx.x * 32 + kloc;
    float s = 0.0f, q = 0.0f;
    for (int b = slice; b < Brows; b += 8) {
        float v = P[(size_t)b * Kcols + k];
        s += v;
        q = fmaf(v, v, q);
    }
    redS[slice][kloc] = s;
    redQ[slice][kloc] = q;
    __syncthreads();
    if (slice == 0) {
        float S = 0.0f, Q = 0.0f;
        #pragma unroll
        for (int i = 0; i < 8; ++i) { S += redS[i][kloc]; Q += redQ[i][kloc]; }
        const float m   = S / (float)Brows;
        const float var = (Q - (float)Brows * m * m) / (float)(Brows - 1);
        const float sd  = sqrtf(fmaxf(var, 0.0f));
        mu[k]   = m;
        rstd[k] = 1.0f / (sd + 1e-8f);
    }
}

// ------------------------------------------------- Epps-Pulley reduction
__global__ __launch_bounds__(256) void sig_ep_kernel(
    const float* __restrict__ P, const float* __restrict__ mu,
    const float* __restrict__ rstd, float* __restrict__ out) {
    const int k   = blockIdx.x;
    const int tid = threadIdx.x;
    const float m = mu[k], r = rstd[k];
    const float DT = 2.0f / (float)NT;

    float sc[NT], ss[NT];
    #pragma unroll
    for (int j = 0; j < NT; ++j) { sc[j] = 0.0f; ss[j] = 0.0f; }

    for (int b = tid; b < Brows; b += 256) {
        const float z = (P[(size_t)b * Kcols + k] - m) * r;
        float sa, ca;
        __sincosf(z * DT, &sa, &ca);
        float c = ca, s = sa;       // cos/sin(1 * z*DT)
        sc[0] += c; ss[0] += s;
        #pragma unroll
        for (int j = 1; j < NT; ++j) {  // rotate by angle z*DT each step
            const float c2 = c * ca - s * sa;
            const float s2 = s * ca + c * sa;
            c = c2; s = s2;
            sc[j] += c; ss[j] += s;
        }
    }

    // wave32 shuffle reduction
    #pragma unroll
    for (int off = 16; off > 0; off >>= 1)
        #pragma unroll
        for (int j = 0; j < NT; ++j) {
            sc[j] += __shfl_down(sc[j], off, 32);
            ss[j] += __shfl_down(ss[j], off, 32);
        }

    __shared__ float red[8][2 * NT];
    const int lane = tid & 31, wv = tid >> 5;
    if (lane == 0)
        #pragma unroll
        for (int j = 0; j < NT; ++j) { red[wv][j] = sc[j]; red[wv][NT + j] = ss[j]; }
    __syncthreads();

    if (tid == 0) {
        float contrib = 0.0f;
        #pragma unroll
        for (int j = 0; j < NT; ++j) {
            float Sc = 0.0f, Ss = 0.0f;
            #pragma unroll
            for (int w = 0; w < 8; ++w) { Sc += red[w][j]; Ss += red[w][NT + j]; }
            const float er  = Sc / (float)Brows;
            const float ei  = Ss / (float)Brows;
            const float t   = (float)(j + 1) * DT;
            const float tcf = expf(-0.5f * t * t);
            const float wj  = (j == 0 || j == NT - 1) ? DT * 0.5f : DT;
            contrib += wj * (er * er + ei * ei - 2.0f * er * tcf + tcf * tcf);
        }
        atomicAdd(out, contrib / (float)Kcols);
    }
}

// ----------------------------------------------------------------- launch
extern "C" void kernel_launch(void* const* d_in, const int* in_sizes, int n_in,
                              void* d_out, int out_size, void* d_ws, size_t ws_size,
                              hipStream_t stream) {
    const float* emb  = (const float*)d_in[0];   // 8192 x 2048
    const float* dirs = (const float*)d_in[1];   // 2048 x 512
    float* out = (float*)d_out;                  // scalar

    float* ws   = (float*)d_ws;
    float* invn = ws;                 // 512
    float* mu   = ws + 512;           // 512
    float* rstd = ws + 1024;          // 512
    float* proj = ws + 2048;          // 8192*512 (16 MB)

    sig_zero_kernel<<<1, 32, 0, stream>>>(out);
    sig_colnorm_kernel<<<Kcols / 32, 256, 0, stream>>>(dirs, invn);
    dim3 grid(Brows / BM, Kcols / BN);   // 64 x 4
    sig_gemm_kernel<<<grid, 256, 0, stream>>>(emb, dirs, invn, proj);
    sig_colstats_kernel<<<Kcols / 32, 256, 0, stream>>>(proj, mu, rstd);
    sig_ep_kernel<<<Kcols, 256, 0, stream>>>(proj, mu, rstd, out);
}
